// V1_43679817400509
// MI455X (gfx1250) — compile-verified, matrix operand order checked by reading, not run on hardware
//
#include <hip/hip_runtime.h>

typedef __attribute__((ext_vector_type(16))) _Float16 v16h;
typedef __attribute__((ext_vector_type(8)))  _Float16 v8h;
typedef __attribute__((ext_vector_type(8)))  float    v8f;
typedef __attribute__((ext_vector_type(4)))  float    v4f;

#define GN    50000
#define GE    500000
#define GH    8
#define GD    16
#define GHID  128
#define GEDIM 64
#define GNET  5
#define SLOPE 0.05f

#define BS_STRIDE 264   // LDS halfs per B column row (max K 256 + 8 pad)

__device__ __forceinline__ int imin(int a, int b) { return a < b ? a : b; }

// ---------------------------------------------------------------------------
// WMMA GEMM: C[M,Nc] = act( concat_K(A0[M,K0], A1[M,K1]) @ B[K,Nc] + bias )
// Block = 8 waves covering 8 consecutive M-tiles at one N-tile; the block's
// B panel (K x 16) is staged once in LDS as f16. Per 32-K step, each lane
// does 4x global_load_b128 (A) + 2x ds_load_b128 (B) + 1 wmma.
// K0, K1 multiples of 32. K = K0+K1 <= 256.
// ---------------------------------------------------------------------------
__global__ __launch_bounds__(256)
void gemm_wmma_f16(const float* __restrict__ A0, int K0,
                   const float* __restrict__ A1, int K1,
                   const float* __restrict__ B, int ldb,
                   const float* __restrict__ bias,
                   float* __restrict__ C, int ldc,
                   int M, int Nc, int act)
{
    __shared__ _Float16 Bs[16 * BS_STRIDE];

    const int K      = K0 + K1;
    const int tilesN = (Nc + 15) >> 4;
    const int tilesM = (M + 15) >> 4;
    const int tn     = blockIdx.x % tilesN;
    const int tm     = (blockIdx.x / tilesN) * 8 + (threadIdx.x >> 5);
    const int lane   = threadIdx.x & 31;
    const int hs     = lane >> 4;    // lane half (selects K sub-range)
    const int rc     = lane & 15;    // A row / B,C col within tile

    // ---- cooperative B panel load: Bs[n][k] = (f16) B[k][tn*16+n] ----
    {
        const int n    = threadIdx.x & 15;
        const int kk0  = threadIdx.x >> 4;               // 0..15
        const int colg = imin(tn * 16 + n, Nc - 1);
        for (int k = kk0; k < K; k += 16)
            Bs[n * BS_STRIDE + k] = (_Float16)B[(size_t)k * ldb + colg];
    }
    __syncthreads();

    if (tm < tilesM) {
        const int row  = imin(tm * 16 + rc, M - 1);      // clamp: garbage rows unstored
        const int col  = tn * 16 + rc;
        const bool cOK = (col < Nc);

        v8f acc = {};
        const _Float16* bsrow = &Bs[rc * BS_STRIDE];

        // two K segments (A0 then A1), contiguous global k index for B
        for (int seg = 0; seg < 2; ++seg) {
            const float* A  = seg ? A1 : A0;
            const int    Ks = seg ? K1 : K0;
            const int    kg0 = seg ? K0 : 0;             // global k base
            const float* arow = A + (size_t)row * Ks;
            for (int ks = 0; ks < Ks; ks += 32) {
                // A fragment: two contiguous 8-float runs -> 4x b128
                const v4f* pa0 = (const v4f*)(arow + ks + hs * 8);
                const v4f* pa1 = (const v4f*)(arow + ks + 16 + hs * 8);
                v4f a0 = pa0[0], a1 = pa0[1];
                v4f a2 = pa1[0], a3 = pa1[1];
                // B fragment: two 8-half runs from LDS -> 2x ds b128
                const int kg = kg0 + ks;
                v8h b0 = *(const v8h*)(bsrow + kg + hs * 8);
                v8h b1 = *(const v8h*)(bsrow + kg + 16 + hs * 8);

                v16h af, bf;
#pragma unroll
                for (int i = 0; i < 4; ++i) {
                    af[i]      = (_Float16)a0[i];
                    af[i + 4]  = (_Float16)a1[i];
                    af[i + 8]  = (_Float16)a2[i];
                    af[i + 12] = (_Float16)a3[i];
                }
#pragma unroll
                for (int i = 0; i < 8; ++i) {
                    bf[i]     = b0[i];
                    bf[i + 8] = b1[i];
                }
                acc = __builtin_amdgcn_wmma_f32_16x16x32_f16(
                          false, af, false, bf, (short)0, acc, false, false);
            }
        }

#pragma unroll
        for (int r = 0; r < 8; ++r) {
            int m = tm * 16 + r + (hs << 3);
            if (m < M && cOK) {
                float v = acc[r];
                if (bias) v += bias[col];
                if (act == 1) v = v > 0.0f ? v : 0.0f;   // relu
                C[(size_t)m * ldc + col] = v;
            }
        }
    }
}

// ---------------------------------------------------------------------------
// small elementwise / edge kernels
// ---------------------------------------------------------------------------
__device__ inline void atomicMaxF32(float* addr, float val) {
    int* ai  = (int*)addr;
    int  old = __float_as_int(*addr);
    while (__int_as_float(old) < val) {
        int assumed = old;
        old = atomicCAS(ai, assumed, __float_as_int(val));
        if (old == assumed) break;
    }
}

__global__ void k_fill(float* p, float v, int total) {
    for (int i = blockIdx.x * blockDim.x + threadIdx.x; i < total;
         i += gridDim.x * blockDim.x) p[i] = v;
}

__global__ void k_copy(float* dst, const float* src, int total) {
    for (int i = blockIdx.x * blockDim.x + threadIdx.x; i < total;
         i += gridDim.x * blockDim.x) dst[i] = src[i];
}

// layer-0 q branch: feat = q[N,1] @ W[1,128]  (outer product)
__global__ void k_outer(const float* __restrict__ q, const float* __restrict__ W,
                        float* __restrict__ feat) {
    for (int i = blockIdx.x * blockDim.x + threadIdx.x; i < GN * GHID;
         i += gridDim.x * blockDim.x)
        feat[i] = q[i >> 7] * W[i & 127];
}

// ea[t][h] = sum_e ae[h][e] * (edge_emb[t] @ We)[h*64+e]   (tiny: 5x8)
__global__ void k_ea(const float* __restrict__ emb, const float* __restrict__ We,
                     const float* __restrict__ ae, float* __restrict__ ea) {
    int tid = threadIdx.x;
    if (tid >= GNET * GH) return;
    int t = tid / GH, h = tid % GH;
    float s = 0.0f;
    for (int e = 0; e < GEDIM; ++e) {
        float ee = 0.0f;
        for (int k = 0; k < GEDIM; ++k)
            ee += emb[t * GEDIM + k] * We[k * (GH * GEDIM) + h * GEDIM + e];
        s += ae[h * GEDIM + e] * ee;
    }
    ea[t * GH + h] = s;
}

// el[n,h] = feat[n,h,:]·al[h,:],  er likewise
__global__ void k_elr(const float* __restrict__ feat, const float* __restrict__ al,
                      const float* __restrict__ ar, float* __restrict__ el,
                      float* __restrict__ er) {
    int i = blockIdx.x * blockDim.x + threadIdx.x;
    if (i >= GN * GH) return;
    int n = i >> 3, h = i & 7;
    float sl = 0.0f, sr = 0.0f;
#pragma unroll
    for (int d = 0; d < GD; ++d) {
        float f = feat[(size_t)n * GHID + h * GD + d];
        sl += f * al[h * GD + d];
        sr += f * ar[h * GD + d];
    }
    el[i] = sl;
    er[i] = sr;
}

// logits = leaky_relu(el[src]+er[dst]+ea[etype]); running segment max into mx
__global__ void k_logits(const float* __restrict__ el, const float* __restrict__ er,
                         const float* __restrict__ ea, const int* __restrict__ src,
                         const int* __restrict__ dst, const int* __restrict__ et,
                         float* __restrict__ alpha, float* __restrict__ mx) {
    for (int i = blockIdx.x * blockDim.x + threadIdx.x; i < GE * GH;
         i += gridDim.x * blockDim.x) {
        int e = i >> 3, h = i & 7;
        float v = el[src[e] * GH + h] + er[dst[e] * GH + h] + ea[et[e] * GH + h];
        v = v > 0.0f ? v : SLOPE * v;
        alpha[i] = v;
        atomicMaxF32(&mx[dst[e] * GH + h], v);
    }
}

// alpha = exp(logit - mx[dst]); den[dst] += alpha
__global__ void k_expsum(const int* __restrict__ dst, const float* __restrict__ mx,
                         float* __restrict__ alpha, float* __restrict__ den) {
    for (int i = blockIdx.x * blockDim.x + threadIdx.x; i < GE * GH;
         i += gridDim.x * blockDim.x) {
        int e = i >> 3, h = i & 7;
        float ex = __expf(alpha[i] - mx[dst[e] * GH + h]);
        alpha[i] = ex;
        atomicAdd(&den[dst[e] * GH + h], ex);
    }
}

// rst[dst] += (alpha/(den[dst]+eps)) * feat[src]   (per edge, 128 feats)
__global__ void k_aggregate(const float* __restrict__ feat,
                            const float* __restrict__ alpha,
                            const float* __restrict__ den,
                            const int* __restrict__ src, const int* __restrict__ dst,
                            float* __restrict__ rst) {
    for (int i = blockIdx.x * blockDim.x + threadIdx.x; i < GE * GHID;
         i += gridDim.x * blockDim.x) {
        int e = i >> 7, j = i & 127, h = j >> 4;
        int s = src[e], d = dst[e];
        float w = alpha[e * GH + h] / (den[d * GH + h] + 1e-9f);
        atomicAdd(&rst[(size_t)d * GHID + j], w * feat[(size_t)s * GHID + j]);
    }
}

__global__ void k_elu(const float* __restrict__ rst, float* __restrict__ out,
                      int total) {
    for (int i = blockIdx.x * blockDim.x + threadIdx.x; i < total;
         i += gridDim.x * blockDim.x) {
        float v = rst[i];
        out[i] = v > 0.0f ? v : (__expf(v) - 1.0f);
    }
}

// ---------------------------------------------------------------------------
// host orchestration
// ---------------------------------------------------------------------------
static void launch_gemm(const float* A0, int K0, const float* A1, int K1,
                        const float* B, int ldb, const float* bias,
                        float* C, int ldc, int M, int Nc, int act,
                        hipStream_t s) {
    int tilesM = (M + 15) / 16, tilesN = (Nc + 15) / 16;
    int blocks = ((tilesM + 7) / 8) * tilesN;  // 8 waves (M tiles) per block
    gemm_wmma_f16<<<blocks, 256, 0, s>>>(A0, K0, A1, K1, B, ldb, bias, C, ldc,
                                         M, Nc, act);
}

// input indices, setup_inputs() insertion order with params leaves flattened
enum {
    I_FEAT0 = 0, I_FEAT1, I_FEAT2, I_Q,
    I_FC0W, I_FC0B, I_FC1W, I_FC1B, I_FC2W, I_FC2B,
    I_G0W, I_G0EMB, I_G0WE, I_G0AL, I_G0AR, I_G0AE,
    I_G1W, I_G1EMB, I_G1WE, I_G1AL, I_G1AR, I_G1AE,
    I_Q0W, I_Q0EMB, I_Q0WE, I_Q0AL, I_Q0AR, I_Q0AE,
    I_Q1W, I_Q1EMB, I_Q1WE, I_Q1AL, I_Q1AR, I_Q1AE,
    I_FUS0W, I_FUS0B, I_FUS1W, I_FUS1B,
    I_SRC, I_DST, I_ETYPE, I_INODES
};

extern "C" void kernel_launch(void* const* d_in, const int* in_sizes, int n_in,
                              void* d_out, int out_size, void* d_ws, size_t ws_size,
                              hipStream_t stream) {
    (void)in_sizes; (void)n_in; (void)out_size; (void)ws_size;
    auto f  = [&](int i) { return (const float*)d_in[i]; };
    auto ii = [&](int i) { return (const int*)d_in[i]; };

    const int* src = ii(I_SRC);
    const int* dst = ii(I_DST);
    const int* et  = ii(I_ETYPE);

    // workspace layout (floats)
    const size_t NF = (size_t)GN * GHID;
    const size_t N8 = (size_t)GN * GH;
    const size_t E8 = (size_t)GE * GH;
    float* ws = (float*)d_ws;
    float* x      = ws;
    float* qcur   = x      + NF;
    float* featx  = qcur   + NF;
    float* featq  = featx  + NF;
    float* rstx   = featq  + NF;
    float* rstq   = rstx   + NF;
    float* qh     = rstq   + NF;
    float* elx    = qh     + NF;
    float* erx    = elx    + N8;
    float* elq    = erx    + N8;
    float* erq    = elq    + N8;
    float* mxx    = erq    + N8;
    float* denx   = mxx    + N8;
    float* mxq    = denx   + N8;
    float* denq   = mxq    + N8;
    float* alphax = denq   + N8;
    float* alphaq = alphax + E8;
    float* eax    = alphaq + E8;
    float* eaq    = eax    + 64;

    const int TB = 256;
    const int GRD = 16384;
    auto blks = [&](long long t) { int b = (int)((t + TB - 1) / TB); return b < GRD ? b : GRD; };

    // ---- input transforms: x = relu(feat_i @ fc_i.w + b), concatenated ----
    launch_gemm(f(I_FEAT0), 128, nullptr, 0, f(I_FC0W), GHID, f(I_FC0B),
                x,                        GHID, 20000, GHID, 1, stream);
    launch_gemm(f(I_FEAT1), 64, nullptr, 0, f(I_FC1W), GHID, f(I_FC1B),
                x + (size_t)20000 * GHID, GHID, 15000, GHID, 1, stream);
    launch_gemm(f(I_FEAT2), 32, nullptr, 0, f(I_FC2W), GHID, f(I_FC2B),
                x + (size_t)35000 * GHID, GHID, 15000, GHID, 1, stream);

    const int gW[2]  = {I_G0W, I_G1W};
    const int gEM[2] = {I_G0EMB, I_G1EMB}, gWE[2] = {I_G0WE, I_G1WE};
    const int gAL[2] = {I_G0AL, I_G1AL},   gAR[2] = {I_G0AR, I_G1AR}, gAE[2] = {I_G0AE, I_G1AE};
    const int qW[2]  = {I_Q0W, I_Q1W};
    const int qEM[2] = {I_Q0EMB, I_Q1EMB}, qWE[2] = {I_Q0WE, I_Q1WE};
    const int qAL[2] = {I_Q0AL, I_Q1AL},   qAR[2] = {I_Q0AR, I_Q1AR}, qAE[2] = {I_Q0AE, I_Q1AE};

    for (int l = 0; l < 2; ++l) {
        // feat projections
        launch_gemm(x, GHID, nullptr, 0, f(gW[l]), GHID, nullptr,
                    featx, GHID, GN, GHID, 0, stream);
        if (l == 0)
            k_outer<<<blks((long long)GN * GHID), TB, 0, stream>>>(f(I_Q), f(qW[0]), featq);
        else
            launch_gemm(qcur, GHID, nullptr, 0, f(qW[1]), GHID, nullptr,
                        featq, GHID, GN, GHID, 0, stream);

        // per-etype edge attention terms (tiny)
        k_ea<<<1, 64, 0, stream>>>(f(gEM[l]), f(gWE[l]), f(gAE[l]), eax);
        k_ea<<<1, 64, 0, stream>>>(f(qEM[l]), f(qWE[l]), f(qAE[l]), eaq);

        // node attention terms
        k_elr<<<(GN * GH + TB - 1) / TB, TB, 0, stream>>>(featx, f(gAL[l]), f(gAR[l]), elx, erx);
        k_elr<<<(GN * GH + TB - 1) / TB, TB, 0, stream>>>(featq, f(qAL[l]), f(qAR[l]), elq, erq);

        // segment softmax
        k_fill<<<blks(N8), TB, 0, stream>>>(mxx, -3.0e38f, (int)N8);
        k_fill<<<blks(N8), TB, 0, stream>>>(mxq, -3.0e38f, (int)N8);
        k_fill<<<blks(N8), TB, 0, stream>>>(denx, 0.0f, (int)N8);
        k_fill<<<blks(N8), TB, 0, stream>>>(denq, 0.0f, (int)N8);
        k_logits<<<blks(E8), TB, 0, stream>>>(elx, erx, eax, src, dst, et, alphax, mxx);
        k_logits<<<blks(E8), TB, 0, stream>>>(elq, erq, eaq, src, dst, et, alphaq, mxq);
        k_expsum<<<blks(E8), TB, 0, stream>>>(dst, mxx, alphax, denx);
        k_expsum<<<blks(E8), TB, 0, stream>>>(dst, mxq, alphaq, denq);

        // aggregation (+ residual for l>0)
        if (l == 0) {
            k_fill<<<blks(NF), TB, 0, stream>>>(rstx, 0.0f, (int)NF);
            k_fill<<<blks(NF), TB, 0, stream>>>(rstq, 0.0f, (int)NF);
        } else {
            k_copy<<<blks(NF), TB, 0, stream>>>(rstx, x, (int)NF);
            k_copy<<<blks(NF), TB, 0, stream>>>(rstq, qcur, (int)NF);
        }
        k_aggregate<<<blks((long long)GE * GHID), TB, 0, stream>>>(featx, alphax, denx, src, dst, rstx);
        k_aggregate<<<blks((long long)GE * GHID), TB, 0, stream>>>(featq, alphaq, denq, src, dst, rstq);

        // elu
        k_elu<<<blks(NF), TB, 0, stream>>>(rstx, x, (int)NF);
        k_elu<<<blks(NF), TB, 0, stream>>>(rstq, qh, (int)NF);

        // fusion: q_next = concat([x, qh]) @ fus.w + b
        if (l == 0)
            launch_gemm(x, GHID, qh, GHID, f(I_FUS0W), GHID, f(I_FUS0B),
                        qcur, GHID, GN, GHID, 1, stream);   // relu
        else
            launch_gemm(x, GHID, qh, GHID, f(I_FUS1W), 2, f(I_FUS1B),
                        (float*)d_out, 2, GN, 2, 0, stream);
    }
}
